// GPT2Attention_79972291051630
// MI455X (gfx1250) — matrix operand order repaired and checked by reading, not compiled, not run on hardware
//
#include <hip/hip_runtime.h>

// ---------------------------------------------------------------------------
// GPT-2 attention block for MI455X (gfx1250, wave32, WMMA f16 16x16x32).
//   qkv = x@Wa + ba ; attention (no mask, scale 1/8) ; out = A@Wp + bp
// All GEMM-like math runs on v_wmma_f32_16x16x32_f16 (f32 accumulate).
// Weights are pre-transposed to [N][K] so both GEMM tiles stage with
// contiguous b128 loads/stores; LDS is double-buffered.
// ---------------------------------------------------------------------------

typedef __attribute__((ext_vector_type(16))) _Float16 v16h;
typedef __attribute__((ext_vector_type(4)))  _Float16 v4h;
typedef __attribute__((ext_vector_type(8)))  float    v8f;
typedef __attribute__((ext_vector_type(4)))  float    f32x4;
typedef __attribute__((ext_vector_type(4)))  unsigned u32x4;

#define EMB   1024
#define NH    16
#define HD    64
#define TSEQ  2048
#define BATCH 4

union V16H { v16h v; u32x4 u[2]; _Float16 h[16]; };

// ---------------------------------------------------------------- f32 -> f16
__global__ __launch_bounds__(256) void cvt_f32_to_f16(
    const float* __restrict__ in, _Float16* __restrict__ out, int n4)
{
    int i = blockIdx.x * blockDim.x + threadIdx.x;
    if (i < n4) {
        f32x4 f = ((const f32x4*)in)[i];
        v4h o;
        o[0] = (_Float16)f[0]; o[1] = (_Float16)f[1];
        o[2] = (_Float16)f[2]; o[3] = (_Float16)f[3];
        ((v4h*)out)[i] = o;
    }
}

// --------------------------------------- f32 [K][N] -> f16 transposed [N][K]
__global__ __launch_bounds__(256) void cvt_transpose_f32_f16(
    const float* __restrict__ in, _Float16* __restrict__ out, int K, int N)
{
    int i  = blockIdx.x * blockDim.x + threadIdx.x;   // over K*N/4
    int nq = N >> 2;
    int k  = i / nq;
    int n4 = (i - k * nq) << 2;
    f32x4 f = *(const f32x4*)(in + (size_t)k * N + n4);
    #pragma unroll
    for (int j = 0; j < 4; j++)
        out[(size_t)(n4 + j) * K + k] = (_Float16)f[j];
}

// -------------------------------------------------- tiled WMMA GEMM (f16,f32 acc)
// C[M][N] = A[M][K] * Bt[N][K]^T. Block 128x128, 256 threads = 8 waves (2x4),
// wave tile 64x32, K-step 32, double-buffered LDS.
// mode 0: outF[gm*N+gn] = acc + bias[gn]                      (projection)
// mode 1: scatter f16 into Q[M][E], K[M][E], Vt[B][H][D][T]   (QKV)
__global__ __launch_bounds__(256) void wmma_gemm_f16(
    const _Float16* __restrict__ A, const _Float16* __restrict__ Bt,
    const float* __restrict__ bias, float* __restrict__ outF,
    _Float16* __restrict__ Qb, _Float16* __restrict__ Kb,
    _Float16* __restrict__ Vtb,
    int M, int N, int Kd, int mode)
{
    __shared__ __align__(16) _Float16 As[2][128 * 32];
    __shared__ __align__(16) _Float16 Bs[2][128 * 32];

    const int tid   = threadIdx.x;
    const int lane  = tid & 31;
    const int wid   = tid >> 5;
    const int lhalf = lane & 15;
    const int hsel  = lane >> 4;          // 0: lanes 0-15, 1: lanes 16-31
    const int wm    = wid >> 2;           // 0..1 -> 64 rows each
    const int wn    = wid & 3;            // 0..3 -> 32 cols each
    const int m0    = blockIdx.y * 128;
    const int n0    = blockIdx.x * 128;

    v8f acc[4][2];
    for (int i = 0; i < 4; i++)
        for (int j = 0; j < 2; j++)
            for (int e = 0; e < 8; e++) acc[i][j][e] = 0.f;

    // staging: thread -> (row sr, 16-half chunk sc); identical pattern for A/Bt
    const int sr = tid >> 1, sc = (tid & 1) * 16;
    const _Float16* aptr = A  + (size_t)(m0 + sr) * Kd + sc;
    const _Float16* bptr = Bt + (size_t)(n0 + sr) * Kd + sc;

    // prologue: stage k-tile 0 into buffer 0
    u32x4 ra0 = ((const u32x4*)aptr)[0], ra1 = ((const u32x4*)aptr)[1];
    u32x4 rb0 = ((const u32x4*)bptr)[0], rb1 = ((const u32x4*)bptr)[1];
    {
        u32x4* ad = (u32x4*)(As[0] + sr * 32 + sc); ad[0] = ra0; ad[1] = ra1;
        u32x4* bd = (u32x4*)(Bs[0] + sr * 32 + sc); bd[0] = rb0; bd[1] = rb1;
    }
    __syncthreads();

    const int KT = Kd >> 5;
    for (int kk = 0; kk < KT; kk++) {
        const int cur = kk & 1;
        if (kk + 1 < KT) {                 // issue next-tile global loads early
            const _Float16* ap = aptr + (size_t)(kk + 1) * 32;
            const _Float16* bp = bptr + (size_t)(kk + 1) * 32;
            ra0 = ((const u32x4*)ap)[0]; ra1 = ((const u32x4*)ap)[1];
            rb0 = ((const u32x4*)bp)[0]; rb1 = ((const u32x4*)bp)[1];
        }

        V16H bf[2];
        #pragma unroll
        for (int nt = 0; nt < 2; nt++) {
            const u32x4* p = (const u32x4*)(Bs[cur] + (wn * 32 + nt * 16 + lhalf) * 32 + hsel * 16);
            bf[nt].u[0] = p[0]; bf[nt].u[1] = p[1];
        }
        #pragma unroll
        for (int mt = 0; mt < 4; mt++) {
            const _Float16* rb = As[cur] + (wm * 64 + mt * 16 + lhalf) * 32 + hsel * 16;
            V16H af;
            af.u[0] = *(const u32x4*)(rb);
            af.u[1] = *(const u32x4*)(rb + 8);
            #pragma unroll
            for (int nt = 0; nt < 2; nt++)
                acc[mt][nt] = __builtin_amdgcn_wmma_f32_16x16x32_f16(
                    false, af.v, false, bf[nt].v, (short)0, acc[mt][nt], false, false);
        }

        if (kk + 1 < KT) {                 // commit next tile into other buffer
            u32x4* ad = (u32x4*)(As[cur ^ 1] + sr * 32 + sc); ad[0] = ra0; ad[1] = ra1;
            u32x4* bd = (u32x4*)(Bs[cur ^ 1] + sr * 32 + sc); bd[0] = rb0; bd[1] = rb1;
        }
        __syncthreads();
    }

    // ---- epilogue (C layout: lane L, vgpr i -> row i+8*hsel, col lhalf) ----
    const int cb = wn * 32;                       // wave column base in block
    float bv[2];
    #pragma unroll
    for (int nt = 0; nt < 2; nt++) bv[nt] = bias[n0 + cb + nt * 16 + lhalf];

    if (mode == 0) {
        #pragma unroll
        for (int mt = 0; mt < 4; mt++)
            #pragma unroll
            for (int nt = 0; nt < 2; nt++)
                #pragma unroll
                for (int i = 0; i < 8; i++) {
                    int gm = m0 + wm * 64 + mt * 16 + i + hsel * 8;
                    int gn = n0 + cb + nt * 16 + lhalf;
                    outF[(size_t)gm * N + gn] = acc[mt][nt][i] + bv[nt];
                }
    } else if (n0 < EMB) {                        // whole block -> Q
        #pragma unroll
        for (int mt = 0; mt < 4; mt++)
            #pragma unroll
            for (int nt = 0; nt < 2; nt++)
                #pragma unroll
                for (int i = 0; i < 8; i++) {
                    int gm = m0 + wm * 64 + mt * 16 + i + hsel * 8;
                    int gn = n0 + cb + nt * 16 + lhalf;
                    Qb[(size_t)gm * EMB + gn] = (_Float16)(acc[mt][nt][i] + bv[nt]);
                }
    } else if (n0 < 2 * EMB) {                    // whole block -> K
        #pragma unroll
        for (int mt = 0; mt < 4; mt++)
            #pragma unroll
            for (int nt = 0; nt < 2; nt++)
                #pragma unroll
                for (int i = 0; i < 8; i++) {
                    int gm = m0 + wm * 64 + mt * 16 + i + hsel * 8;
                    int gn = n0 - EMB + cb + nt * 16 + lhalf;
                    Kb[(size_t)gm * EMB + gn] = (_Float16)(acc[mt][nt][i] + bv[nt]);
                }
    } else {                                      // whole block -> V transposed
        #pragma unroll
        for (int mt = 0; mt < 4; mt++)
            #pragma unroll
            for (int nt = 0; nt < 2; nt++) {
                int c  = n0 - 2 * EMB + cb + nt * 16 + lhalf;
                int hh = c >> 6, d = c & 63;
                #pragma unroll
                for (int i = 0; i < 8; i++) {
                    int gm = m0 + wm * 64 + mt * 16 + i + hsel * 8;
                    int bb = gm >> 11, t = gm & (TSEQ - 1);
                    Vtb[((size_t)(bb * NH + hh) * HD + d) * TSEQ + t] =
                        (_Float16)(acc[mt][nt][i] + bv[nt]);
                }
            }
    }
}

// ---------------------------------------------------- flash attention (f16 WMMA)
// One wave = 16 queries of one (b,h). Computes S^T = K*Q^T so softmax is
// lane-local and P^T maps lane-for-lane into the A operand of O = P*V.
// V is pre-transposed (Vt[b][h][d][t]) so the O-WMMA B operand is contiguous.
// No LDS, no barriers; 8 WMMAs per 32-key chunk.
__global__ __launch_bounds__(256) void flash_attn_f16(
    const _Float16* __restrict__ Qb, const _Float16* __restrict__ Kb,
    const _Float16* __restrict__ Vtb, _Float16* __restrict__ Ah)
{
    const int lane  = threadIdx.x & 31;
    const int wave  = threadIdx.x >> 5;
    const int lhalf = lane & 15;
    const int hsel  = lane >> 4;
    const int qt = blockIdx.x & 15;          // 16 query tiles of 128 per (b,h)
    const int bh = blockIdx.x >> 4;
    const int h  = bh & (NH - 1);
    const int b  = bh >> 4;
    const int qbase = qt * 128 + wave * 16;

    // Q^T fragments (B operand): lane = query lhalf, contiguous d per half
    V16H qf[2];
    {
        const _Float16* qrow = Qb + (size_t)(b * TSEQ + qbase + lhalf) * EMB + h * HD;
        #pragma unroll
        for (int c = 0; c < 2; c++) {
            const u32x4* p = (const u32x4*)(qrow + c * 32 + hsel * 16);
            qf[c].u[0] = p[0]; qf[c].u[1] = p[1];
        }
    }

    v8f o[4];
    for (int dt = 0; dt < 4; dt++)
        for (int e = 0; e < 8; e++) o[dt][e] = 0.f;
    float m = -1e30f, l = 0.f;

    const _Float16* kbh = Kb + (size_t)(b * TSEQ) * EMB + h * HD;
    const _Float16* vbh = Vtb + (size_t)(b * NH + h) * HD * TSEQ;

    for (int kc = 0; kc < TSEQ / 32; kc++) {
        // S^T = K * Q^T : two 16(key)x16(query) tiles, k-dim = d (2 x 32)
        v8f s[2];
        for (int e = 0; e < 8; e++) { s[0][e] = 0.f; s[1][e] = 0.f; }
        #pragma unroll
        for (int kt = 0; kt < 2; kt++) {
            #pragma unroll
            for (int dc = 0; dc < 2; dc++) {
                const _Float16* kr =
                    kbh + (size_t)(kc * 32 + kt * 16 + lhalf) * EMB + dc * 32 + hsel * 16;
                V16H af;
                af.u[0] = *(const u32x4*)(kr);
                af.u[1] = *(const u32x4*)(kr + 8);
                s[kt] = __builtin_amdgcn_wmma_f32_16x16x32_f16(
                    false, af.v, false, qf[dc].v, (short)0, s[kt], false, false);
            }
        }

        // online softmax: lane owns query lhalf; partner lane (xor 16) holds
        // the other 16 of this chunk's 32 keys.
        float cm = -1e30f;
        #pragma unroll
        for (int i = 0; i < 8; i++) {
            cm = fmaxf(cm, s[0][i]);
            cm = fmaxf(cm, s[1][i]);
        }
        cm *= 0.125f;                                   // 1/sqrt(64)
        cm = fmaxf(cm, __shfl_xor(cm, 16, 32));
        float mn   = fmaxf(m, cm);
        float corr = __expf(m - mn);
        float p0[8], p1[8], rs = 0.f;
        #pragma unroll
        for (int i = 0; i < 8; i++) {
            p0[i] = __expf(s[0][i] * 0.125f - mn); rs += p0[i];
            p1[i] = __expf(s[1][i] * 0.125f - mn); rs += p1[i];
        }
        rs += __shfl_xor(rs, 16, 32);
        l = l * corr + rs;
        m = mn;

        // rescale O accumulators: vgpr i holds query i+8*hsel -> broadcast corr
        float cq[8];
        #pragma unroll
        for (int i = 0; i < 8; i++) cq[i] = __shfl(corr, i + hsel * 8, 32);
        #pragma unroll
        for (int dt = 0; dt < 4; dt++)
            #pragma unroll
            for (int i = 0; i < 8; i++) o[dt][i] *= cq[i];

        // P (A operand, 16 queries x 32 keys) — lane-local repack of P^T tiles
        V16H pa;
        #pragma unroll
        for (int i = 0; i < 8; i++) {
            pa.h[i]     = (_Float16)p0[i];
            pa.h[8 + i] = (_Float16)p1[i];
        }

        // O += P * V : B operand from Vt[d][t] (keys contiguous)
        #pragma unroll
        for (int dt = 0; dt < 4; dt++) {
            const _Float16* vr = vbh + (size_t)(dt * 16 + lhalf) * TSEQ + kc * 32 + hsel * 8;
            V16H vf;
            vf.u[0] = ((const u32x4*)vr)[0];
            vf.u[1] = ((const u32x4*)vr)[2];
            o[dt] = __builtin_amdgcn_wmma_f32_16x16x32_f16(
                false, pa.v, false, vf.v, (short)0, o[dt], false, false);
        }
    }

    // normalize and store merged-head output [B*T][E] as f16
    float li[8];
    #pragma unroll
    for (int i = 0; i < 8; i++) li[i] = 1.f / __shfl(l, i + hsel * 8, 32);
    #pragma unroll
    for (int dt = 0; dt < 4; dt++) {
        #pragma unroll
        for (int i = 0; i < 8; i++) {
            size_t row = (size_t)(b * TSEQ + qbase + i + hsel * 8);
            int    col = h * HD + dt * 16 + lhalf;
            Ah[row * EMB + col] = (_Float16)(o[dt][i] * li[i]);
        }
    }
}

// ---------------------------------------------------------------------------
extern "C" void kernel_launch(void* const* d_in, const int* in_sizes, int n_in,
                              void* d_out, int out_size, void* d_ws, size_t ws_size,
                              hipStream_t stream)
{
    const float* x      = (const float*)d_in[0];   // [4,2048,1024]
    const float* w_attn = (const float*)d_in[1];   // [1024,3072]
    const float* b_attn = (const float*)d_in[2];   // [3072]
    const float* w_proj = (const float*)d_in[3];   // [1024,1024]
    const float* b_proj = (const float*)d_in[4];   // [1024]
    float* out = (float*)d_out;                    // [4,2048,1024]

    const size_t NX  = (size_t)BATCH * TSEQ * EMB;     // 8388608
    const size_t NWA = (size_t)EMB * 3 * EMB;          // 3145728
    const size_t NWP = (size_t)EMB * EMB;              // 1048576

    _Float16* base = (_Float16*)d_ws;
    _Float16* x_h  = base;                 // also reused as attention output Ah
    _Float16* wa_t = x_h  + NX;            // w_attn transposed [3E][E] f16
    _Float16* wp_t = wa_t + NWA;           // w_proj transposed [E][E] f16
    _Float16* Qb   = wp_t + NWP;
    _Float16* Kb   = Qb + NX;
    _Float16* Vtb  = Kb + NX;
    _Float16* Ah   = x_h;                  // alias: x_h dead after QKV GEMM

    // 1. convert x to f16; transpose-convert weights to f16 [N][K]
    cvt_f32_to_f16<<<(int)(NX / 4 / 256), 256, 0, stream>>>(x, x_h, (int)(NX / 4));
    cvt_transpose_f32_f16<<<(int)(NWA / 4 / 256), 256, 0, stream>>>(
        w_attn, wa_t, EMB, 3 * EMB);
    cvt_transpose_f32_f16<<<(int)(NWP / 4 / 256), 256, 0, stream>>>(
        w_proj, wp_t, EMB, EMB);

    // 2. QKV GEMM: [8192,1024] x [1024,3072] -> Q/K f16 + V transposed
    {
        dim3 grid(3 * EMB / 128, BATCH * TSEQ / 128);
        wmma_gemm_f16<<<grid, 256, 0, stream>>>(
            x_h, wa_t, b_attn, nullptr, Qb, Kb, Vtb,
            BATCH * TSEQ, 3 * EMB, EMB, 1);
    }

    // 3. attention: 1024 blocks x 8 waves, 16 queries/wave
    flash_attn_f16<<<BATCH * NH * (TSEQ / 128), 256, 0, stream>>>(Qb, Kb, Vtb, Ah);

    // 4. output projection: [8192,1024] x [1024,1024] -> f32 out + bias
    {
        dim3 grid(EMB / 128, BATCH * TSEQ / 128);
        wmma_gemm_f16<<<grid, 256, 0, stream>>>(
            Ah, wp_t, b_proj, out, nullptr, nullptr, nullptr,
            BATCH * TSEQ, EMB, EMB, 0);
    }
}
